// Moe_78984448573477
// MI455X (gfx1250) — matrix operand order repaired and checked by reading, not compiled, not run on hardware
//
#include <hip/hip_runtime.h>
#include <hip/hip_bf16.h>

// ---------------- problem constants ----------------
#define Nn 8192   // B*T tokens
#define Dd 1024   // model dim
#define Ff 4096   // ffn hidden
#define Ee 8      // experts
#define Kk 2      // top-k
#define MT 32     // token tile = two 16-row WMMA A-tiles sharing B fragments
#define FC 512    // F chunk held in LDS per phase
#define NCHUNK (Ff / FC)

typedef __hip_bfloat16 bf16_t;

typedef __attribute__((ext_vector_type(16))) __bf16 v16bf;
typedef __attribute__((ext_vector_type(8)))  __bf16 v8bf;
typedef __attribute__((ext_vector_type(8)))  float  v8f;

__device__ __forceinline__ v8f wmma_bf16(v16bf a, v16bf b, v8f c) {
  // (neg_a, A, neg_b, B, c_mod, C, reuse_a, reuse_b)
  return __builtin_amdgcn_wmma_f32_16x16x32_bf16(false, a, false, b, (short)0, c,
                                                 false, false);
}

// A-matrix fragment (16x32 bf16, row-major source, per-lane row = lane&15).
// lanes 0-15: halves 0-15 = K k0+0..15
// lanes16-31: halves 0-15 = K k0+16..31
__device__ __forceinline__ v16bf load_af(const __bf16* rowp, int k0, int hi) {
  const v8bf lo = *reinterpret_cast<const v8bf*>(rowp + k0 + (hi ? 16 : 0));
  const v8bf hh = *reinterpret_cast<const v8bf*>(rowp + k0 + (hi ? 24 : 8));
  v16bf a;
#pragma unroll
  for (int i = 0; i < 8; ++i) { a[i] = lo[i]; a[i + 8] = hh[i]; }
  return a;
}

// B-matrix fragment (32x16 bf16): lane holds column n = lane&15,
// 16 consecutive K halves starting at k0 (+16 for lanes 16-31).
__device__ __forceinline__ v16bf load_bf(const __bf16* p) {
  const v8bf lo = *reinterpret_cast<const v8bf*>(p);
  const v8bf hh = *reinterpret_cast<const v8bf*>(p + 8);
  v16bf b;
#pragma unroll
  for (int i = 0; i < 8; ++i) { b[i] = lo[i]; b[i + 8] = hh[i]; }
  return b;
}

// ---------------- prep kernels ----------------

__global__ void conv_x_kernel(const float* __restrict__ x, bf16_t* __restrict__ xb) {
  int i = (blockIdx.x * blockDim.x + threadIdx.x) * 4;
  if (i >= Nn * Dd) return;
  float4 v = *reinterpret_cast<const float4*>(x + i);
  xb[i + 0] = __float2bfloat16(v.x);
  xb[i + 1] = __float2bfloat16(v.y);
  xb[i + 2] = __float2bfloat16(v.z);
  xb[i + 3] = __float2bfloat16(v.w);
}

// in: [E][R][C] f32 -> out: [E][C][R] bf16 (tiled transpose via LDS)
__global__ void transpose_to_bf16(const float* __restrict__ in,
                                  bf16_t* __restrict__ out, int R, int C) {
  __shared__ float tile[32][33];
  const int e = blockIdx.z;
  in  += (size_t)e * R * C;
  out += (size_t)e * R * C;
  const int c0 = blockIdx.x * 32, r0 = blockIdx.y * 32;
  const int tx = threadIdx.x, ty = threadIdx.y;  // 32 x 8
#pragma unroll
  for (int i = 0; i < 32; i += 8) {
    int r = r0 + ty + i, c = c0 + tx;
    tile[ty + i][tx] = (r < R && c < C) ? in[(size_t)r * C + c] : 0.f;
  }
  __syncthreads();
#pragma unroll
  for (int i = 0; i < 32; i += 8) {
    int c = c0 + ty + i, r = r0 + tx;
    if (c < C && r < R) out[(size_t)c * R + r] = __float2bfloat16(tile[tx][ty + i]);
  }
}

__global__ void gate_logits_kernel(const float* __restrict__ x,
                                   const float* __restrict__ Wg,
                                   const float* __restrict__ bg,
                                   float* __restrict__ logits) {
  int idx = blockIdx.x * blockDim.x + threadIdx.x;
  if (idx >= Nn * Ee) return;
  int t = idx / Ee, e = idx % Ee;
  float s = bg[e];
  const float* xr = x + (size_t)t * Dd;
  for (int d = 0; d < Dd; ++d) s = fmaf(xr[d], Wg[d * Ee + e], s);
  logits[idx] = s;
}

__global__ void topk_softmax_kernel(const float* __restrict__ logits,
                                    int* __restrict__ exp_id,
                                    float* __restrict__ exp_w) {
  int t = blockIdx.x * blockDim.x + threadIdx.x;
  if (t >= Nn) return;
  const float* L = logits + (size_t)t * Ee;
  int i1 = 0; float v1 = L[0];
#pragma unroll
  for (int e = 1; e < Ee; ++e) if (L[e] > v1) { v1 = L[e]; i1 = e; }
  int i2 = -1; float v2 = -3.4e38f;
#pragma unroll
  for (int e = 0; e < Ee; ++e) {
    if (e == i1) continue;
    if (L[e] > v2) { v2 = L[e]; i2 = e; }
  }
  float p2 = expf(v2 - v1);
  float inv = 1.f / (1.f + p2);
  exp_id[t * 2 + 0] = i1; exp_w[t * 2 + 0] = inv;
  exp_id[t * 2 + 1] = i2; exp_w[t * 2 + 1] = p2 * inv;
}

// deterministic, order-preserving bucket build: one thread per expert
__global__ void build_buckets_kernel(const int* __restrict__ exp_id,
                                     const float* __restrict__ exp_w,
                                     int* __restrict__ btok,
                                     float* __restrict__ bw,
                                     int* __restrict__ counts) {
  int e = threadIdx.x;
  if (e >= Ee) return;
  int cnt = 0;
  for (int t = 0; t < Nn; ++t) {
#pragma unroll
    for (int k = 0; k < Kk; ++k) {
      if (exp_id[t * Kk + k] == e) {
        btok[e * Nn + cnt] = t;
        bw[e * Nn + cnt] = exp_w[t * Kk + k];
        ++cnt;
      }
    }
  }
  counts[e] = cnt;
}

__global__ void zero_out_kernel(float* __restrict__ out, int n4) {
  int i = blockIdx.x * blockDim.x + threadIdx.x;
  if (i < n4) reinterpret_cast<float4*>(out)[i] = make_float4(0.f, 0.f, 0.f, 0.f);
}

// ---------------- fused expert MLP (WMMA bf16, M=32 B-reuse) ----------------
__global__ __launch_bounds__(256, 1) void moe_expert_mlp(
    const bf16_t* __restrict__ xb_, const bf16_t* __restrict__ W1t_,
    const bf16_t* __restrict__ W2t_, const float* __restrict__ b1,
    const float* __restrict__ b2, const int* __restrict__ counts,
    const int* __restrict__ btok, const float* __restrict__ bw,
    float* __restrict__ out) {
  const int e = blockIdx.y;
  const int tile = blockIdx.x;
  const int cnt = counts[e];
  if (tile * MT >= cnt) return;

  const __bf16* xb  = reinterpret_cast<const __bf16*>(xb_);
  const __bf16* W1t = reinterpret_cast<const __bf16*>(W1t_) + (size_t)e * Ff * Dd; // [F][D]
  const __bf16* W2t = reinterpret_cast<const __bf16*>(W2t_) + (size_t)e * Dd * Ff; // [D][F]
  const float* b1e = b1 + (size_t)e * Ff;
  const float* b2e = b2 + (size_t)e * Dd;

  __shared__ __bf16 xs[MT][Dd + 8];   // 66.0 KB (x tile, row-major)
  __shared__ __bf16 hs[MT][FC + 8];   // 33.3 KB (gelu(h) chunk, row-major)
  __shared__ int    s_tok[MT];
  __shared__ float  s_w[MT];

  const int tid  = threadIdx.x;
  const int lane = tid & 31;
  const int wave = tid >> 5;          // 8 waves
  const int m    = lane & 15;
  const int hi   = lane >> 4;

  if (tid < MT) {
    int g = tile * MT + tid;
    if (g < cnt) { s_tok[tid] = btok[e * Nn + g]; s_w[tid] = bw[e * Nn + g]; }
    else         { s_tok[tid] = -1;               s_w[tid] = 0.f; }
  }
  __syncthreads();

  // gather x rows into LDS (padded rows -> zeros): 32 rows, 8 threads/row
  {
    const int row = tid >> 3;
    const int seg = tid & 7;          // 128 halves per segment
    const int tok = s_tok[row];
#pragma unroll
    for (int i = 0; i < 16; ++i) {
      v8bf v;
      if (tok >= 0) {
        v = *reinterpret_cast<const v8bf*>(xb + (size_t)tok * Dd + seg * 128 + i * 8);
      } else {
#pragma unroll
        for (int q = 0; q < 8; ++q) v[q] = (__bf16)0.f;
      }
      *reinterpret_cast<v8bf*>(&xs[row][seg * 128 + i * 8]) = v;
    }
  }
  __syncthreads();

  // y accumulators: 2 row-tiles x 8 col-tiles; wave owns d = (wave*8+j)*16 + n
  v8f acc2[2][8];
#pragma unroll
  for (int t = 0; t < 2; ++t)
#pragma unroll
    for (int j = 0; j < 8; ++j)
#pragma unroll
      for (int q = 0; q < 8; ++q) acc2[t][j][q] = 0.f;

  const __bf16* xrow0 = &xs[m][0];
  const __bf16* xrow1 = &xs[16 + m][0];
  const __bf16* hrow0 = &hs[m][0];
  const __bf16* hrow1 = &hs[16 + m][0];

  for (int ch = 0; ch < NCHUNK; ++ch) {
    const int f0 = ch * FC;

    // ---- phase 1: h = gelu(x @ W1 + b1) for this F chunk ----
    v8f acc1[2][4];
#pragma unroll
    for (int t = 0; t < 2; ++t)
#pragma unroll
      for (int i = 0; i < 4; ++i)
#pragma unroll
        for (int q = 0; q < 8; ++q) acc1[t][i][q] = 0.f;

    for (int k0 = 0; k0 < Dd; k0 += 32) {
      v16bf a0 = load_af(xrow0, k0, hi);
      v16bf a1 = load_af(xrow1, k0, hi);
#pragma unroll
      for (int i = 0; i < 4; ++i) {
        const int fg = f0 + (wave * 4 + i) * 16 + m;
        v16bf b = load_bf(W1t + (size_t)fg * Dd + k0 + (hi ? 16 : 0));
        acc1[0][i] = wmma_bf16(a0, b, acc1[0][i]);
        acc1[1][i] = wmma_bf16(a1, b, acc1[1][i]);
      }
    }
    // exact-erf GELU epilogue -> bf16 in LDS
#pragma unroll
    for (int i = 0; i < 4; ++i) {
      const int fl = (wave * 4 + i) * 16 + m;
      const float bias = b1e[f0 + fl];
#pragma unroll
      for (int t = 0; t < 2; ++t) {
#pragma unroll
        for (int v = 0; v < 8; ++v) {
          const int r = v + (hi ? 8 : 0) + t * 16;
          float val = acc1[t][i][v] + bias;
          float g = 0.5f * val * (1.0f + erff(val * 0.70710678118654752f));
          hs[r][fl] = (__bf16)g;
        }
      }
    }
    __syncthreads();

    // ---- phase 2: y += h @ W2 over this F chunk ----
    for (int k0 = 0; k0 < FC; k0 += 32) {
      v16bf a0 = load_af(hrow0, k0, hi);
      v16bf a1 = load_af(hrow1, k0, hi);
      const int kg = f0 + k0 + (hi ? 16 : 0);
#pragma unroll
      for (int j = 0; j < 8; ++j) {
        const int d = (wave * 8 + j) * 16 + m;
        v16bf b = load_bf(W2t + (size_t)d * Ff + kg);
        acc2[0][j] = wmma_bf16(a0, b, acc2[0][j]);
        acc2[1][j] = wmma_bf16(a1, b, acc2[1][j]);
      }
    }
    __syncthreads();
  }

  // ---- out[tok] += w * (y + b2); exactly 2 adds/element -> deterministic ----
#pragma unroll
  for (int j = 0; j < 8; ++j) {
    const int d = (wave * 8 + j) * 16 + m;
    const float bias2 = b2e[d];
#pragma unroll
    for (int t = 0; t < 2; ++t) {
#pragma unroll
      for (int v = 0; v < 8; ++v) {
        const int r = v + (hi ? 8 : 0) + t * 16;
        const int tok = s_tok[r];
        if (tok >= 0)
          atomicAdd(out + (size_t)tok * Dd + d, s_w[r] * (acc2[t][j][v] + bias2));
      }
    }
  }
}

// ---------------- host launcher ----------------
extern "C" void kernel_launch(void* const* d_in, const int* in_sizes, int n_in,
                              void* d_out, int out_size, void* d_ws, size_t ws_size,
                              hipStream_t stream) {
  const float* x  = (const float*)d_in[0];
  const float* Wg = (const float*)d_in[1];
  const float* bg = (const float*)d_in[2];
  const float* W1 = (const float*)d_in[3];
  const float* b1 = (const float*)d_in[4];
  const float* W2 = (const float*)d_in[5];
  const float* b2 = (const float*)d_in[6];
  float* out = (float*)d_out;

  char* ws = (char*)d_ws;
  size_t used = 0;
  auto alloc = [&](size_t bytes) -> void* {
    void* p = ws + used;
    used += (bytes + 255) & ~(size_t)255;
    return p;
  };
  bf16_t* xb    = (bf16_t*)alloc((size_t)Nn * Dd * 2);
  bf16_t* W1t   = (bf16_t*)alloc((size_t)Ee * Dd * Ff * 2);
  bf16_t* W2t   = (bf16_t*)alloc((size_t)Ee * Ff * Dd * 2);
  float* logits = (float*)alloc((size_t)Nn * Ee * 4);
  int*   exp_id = (int*)alloc((size_t)Nn * Kk * 4);
  float* exp_w  = (float*)alloc((size_t)Nn * Kk * 4);
  int*   counts = (int*)alloc((size_t)Ee * 4);
  int*   btok   = (int*)alloc((size_t)Ee * Nn * 4);
  float* bwgt   = (float*)alloc((size_t)Ee * Nn * 4);
  if (used > ws_size) return;  // workspace too small; bail safely

  conv_x_kernel<<<(Nn * Dd / 4 + 255) / 256, 256, 0, stream>>>(x, xb);

  dim3 tb(32, 8);
  transpose_to_bf16<<<dim3(Ff / 32, Dd / 32, Ee), tb, 0, stream>>>(W1, W1t, Dd, Ff);
  transpose_to_bf16<<<dim3(Dd / 32, Ff / 32, Ee), tb, 0, stream>>>(W2, W2t, Ff, Dd);

  gate_logits_kernel<<<(Nn * Ee + 255) / 256, 256, 0, stream>>>(x, Wg, bg, logits);
  topk_softmax_kernel<<<(Nn + 255) / 256, 256, 0, stream>>>(logits, exp_id, exp_w);
  build_buckets_kernel<<<1, 32, 0, stream>>>(exp_id, exp_w, btok, bwgt, counts);

  zero_out_kernel<<<(Nn * Dd / 4 + 255) / 256, 256, 0, stream>>>(out, Nn * Dd / 4);

  moe_expert_mlp<<<dim3(Nn / MT, Ee), 256, 0, stream>>>(
      xb, W1t, W2t, b1, b2, counts, btok, bwgt, out);
}
